// _LocalConditioner_71476845740647
// MI455X (gfx1250) — matrix operand order repaired and checked by reading, not compiled, not run on hardware
//
#include <hip/hip_runtime.h>
#include <stdint.h>

// ---------------- problem constants ----------------
#define SEQ    131072
#define TD     8
#define CD     16
#define NROWS  65536          // N = SEQ/2 target sites
#define FEAT_P 64             // 56 features padded to 64
#define WIDTH  512
#define OUTC   16
#define MBLK   32             // rows per workgroup (2 WMMA M-tiles)

typedef __attribute__((ext_vector_type(16))) __bf16 v16bf;
typedef __attribute__((ext_vector_type(8)))  __bf16 v8bf;
typedef __attribute__((ext_vector_type(8)))  float  v8f;

// f32 -> bf16 round-to-nearest-even (pure integer ops; always compiles)
__device__ __forceinline__ __bf16 f2bf(float f) {
  union { float f; uint32_t u; } v; v.f = f;
  uint32_t r = (v.u + 0x7FFFu + ((v.u >> 16) & 1u)) >> 16;
  union { uint16_t u; __bf16 b; } o; o.u = (uint16_t)r;
  return o.b;
}

__device__ __forceinline__ int hrank(int n) { return (n & 7) - 1; } // hidden rank

// ---------------- prep: feature matrix [N][64] bf16 ----------------
__global__ __launch_bounds__(256) void prep_feats(
    const float* __restrict__ xt, const float* __restrict__ xf,
    const float* __restrict__ cond, __bf16* __restrict__ F) {
  int idx = blockIdx.x * 256 + threadIdx.x;
  if (idx >= NROWS * FEAT_P) return;
  int r = idx >> 6, c = idx & 63;
  int t = r * 2;                       // TARGET_IX = 0,2,4,...
  float v = 0.f;
  if (c < TD) {
    v = xt[r * TD + c];
  } else if (c < TD + 32) {
    int o = (c - TD) >> 3, d = (c - TD) & 7;
    int off = (o < 2) ? (o - 2) : (o - 1);      // offsets {-2,-1,1,2}
    int s = t + off;                            // edge-pad == clamp
    s = s < 0 ? 0 : (s > SEQ - 1 ? SEQ - 1 : s);
    v = xf[s * TD + d];
  } else if (c < TD + 32 + CD) {
    v = cond[t * CD + (c - TD - 32)];
  } // else zero pad (cols 56..63)
  F[idx] = f2bf(v);
}

// ---------------- prep: masked weights in B-fragment order ----------------
// Fragment (kt,nt): 32 lanes x 16 bf16 contiguous per lane.
// Element e of lane l maps to K = kt*32 + e + 16*(l>>4), n = nt*16 + (l&15).
#define NW0 (2  * 32 * 512)   // KT=2,  NT=32
#define NW1 (16 * 32 * 512)   // KT=16, NT=32
#define NW2 (16 * 1  * 512)   // KT=16, NT=1
__global__ __launch_bounds__(256) void prep_weights(
    const float* __restrict__ W0, const float* __restrict__ W1,
    const float* __restrict__ W2,
    __bf16* __restrict__ W0f, __bf16* __restrict__ W1f, __bf16* __restrict__ W2f) {
  int idx = blockIdx.x * 256 + threadIdx.x;
  if (idx >= NW0 + NW1 + NW2) return;

  const float* W; __bf16* Wf; int NT, local, which;
  if (idx < NW0)            { which = 0; W = W0; Wf = W0f; NT = 32; local = idx; }
  else if (idx < NW0 + NW1) { which = 1; W = W1; Wf = W1f; NT = 32; local = idx - NW0; }
  else                      { which = 2; W = W2; Wf = W2f; NT = 1;  local = idx - NW0 - NW1; }

  int frag = local >> 9;            // /512
  int rem  = local & 511;
  int lane = rem >> 4, e = rem & 15;
  int kt = frag / NT, nt = frag % NT;
  int K = kt * 32 + e + 16 * (lane >> 4);
  int n = nt * 16 + (lane & 15);

  float v = 0.f;
  if (which == 0) {
    // W0 is [512 x 56]; in_ranks: 0..7 then -1; mask: K<8 -> hr(n)>=K, 8<=K<56 -> 1
    if (K < 56) {
      bool m = (K < TD) ? (hrank(n) >= K) : true;
      v = m ? W0[n * 56 + K] : 0.f;
    }
  } else if (which == 1) {
    // W1 [512 x 512]; mask: hr(n) >= hr(K)
    v = (hrank(n) >= hrank(K)) ? W1[n * 512 + K] : 0.f;
  } else {
    // W2 [16 x 512]; out_rank(n)=n/2; mask: out_rank > hr(K)
    v = ((n >> 1) > hrank(K)) ? W2[n * 512 + K] : 0.f;
  }
  Wf[local] = f2bf(v);
}

// ---------------- main fused MLP ----------------
__device__ __forceinline__ v16bf loadA_lds(const __bf16* base, int stride,
                                           int mtile, int k0, int mlo, int khg) {
  // A 16x32 bf16 fragment: lane = row (lo/hi K-half by lane group);
  // one contiguous 16-element run (K = 16*khg + e) -> two ds_load_b128
  const __bf16* p = base + (mtile * 16 + mlo) * stride + k0 + khg * 16;
  v8bf lo = *(const v8bf*)p;
  v8bf hi = *(const v8bf*)(p + 8);
  v16bf a;
#pragma unroll
  for (int e = 0; e < 8; ++e) { a[e] = lo[e]; a[8 + e] = hi[e]; }
  return a;
}

__device__ __forceinline__ v16bf loadB_frag(const __bf16* wf, int frag, int lane) {
  // pre-swizzled: 16 contiguous bf16 per lane -> two global_load_b128
  const __bf16* p = wf + ((frag * 32 + lane) << 4);
  v8bf lo = *(const v8bf*)p;
  v8bf hi = *(const v8bf*)(p + 8);
  v16bf b;
#pragma unroll
  for (int e = 0; e < 8; ++e) { b[e] = lo[e]; b[8 + e] = hi[e]; }
  return b;
}

#define WMMA_BF16(A, B, C) \
  __builtin_amdgcn_wmma_f32_16x16x32_bf16(false, (A), false, (B), (short)0, (C), false, false)

__global__ __launch_bounds__(256) void made_mlp(
    const __bf16* __restrict__ F,
    const __bf16* __restrict__ W0f, const __bf16* __restrict__ W1f,
    const __bf16* __restrict__ W2f,
    const float* __restrict__ b0, const float* __restrict__ b1,
    const float* __restrict__ b2, float* __restrict__ out) {
  __shared__ __align__(16) __bf16 lds0[MBLK * WIDTH];  // feats (stride 64), later h1 (stride 512)
  __shared__ __align__(16) __bf16 lds1[MBLK * WIDTH];  // h0 (stride 512)

  const int tid  = threadIdx.x;
  const int lane = tid & 31;
  const int w    = tid >> 5;          // 8 waves: wave w owns n-tiles 4w..4w+3
  const int mlo  = lane & 15;
  const int khg  = lane >> 4;
  const int rowBase = blockIdx.x * MBLK;

  // stage 32x64 bf16 features into LDS (16B per thread x2, coalesced)
  {
    const v8bf* src = (const v8bf*)(F + (size_t)rowBase * FEAT_P);
    v8bf* dst = (v8bf*)lds0;
    dst[tid]       = src[tid];
    dst[tid + 256] = src[tid + 256];
  }
  __syncthreads();

  // ---------------- layer 0: [32x64] x [64x512] ----------------
  v8f acc[2][4];
#pragma unroll
  for (int m = 0; m < 2; ++m)
#pragma unroll
    for (int i = 0; i < 4; ++i) acc[m][i] = (v8f)0.f;

#pragma unroll
  for (int kt = 0; kt < 2; ++kt) {
    v16bf a0 = loadA_lds(lds0, FEAT_P, 0, kt * 32, mlo, khg);
    v16bf a1 = loadA_lds(lds0, FEAT_P, 1, kt * 32, mlo, khg);
#pragma unroll
    for (int i = 0; i < 4; ++i) {
      v16bf bf = loadB_frag(W0f, kt * 32 + (w * 4 + i), lane);
      acc[0][i] = WMMA_BF16(a0, bf, acc[0][i]);
      acc[1][i] = WMMA_BF16(a1, bf, acc[1][i]);
    }
  }
#pragma unroll
  for (int i = 0; i < 4; ++i) {
    int col = (w * 4 + i) * 16 + mlo;
    float bias = b0[col];
#pragma unroll
    for (int v = 0; v < 8; ++v) {
      float x = acc[0][i][v] + bias; x = x > 0.f ? x : 0.f;
      lds1[(v + 8 * khg) * WIDTH + col] = f2bf(x);
      float y = acc[1][i][v] + bias; y = y > 0.f ? y : 0.f;
      lds1[(16 + v + 8 * khg) * WIDTH + col] = f2bf(y);
    }
  }
  __syncthreads();

  // ---------------- layer 1: [32x512] x [512x512] ----------------
#pragma unroll
  for (int m = 0; m < 2; ++m)
#pragma unroll
    for (int i = 0; i < 4; ++i) acc[m][i] = (v8f)0.f;

  for (int kt = 0; kt < 16; ++kt) {
    v16bf a0 = loadA_lds(lds1, WIDTH, 0, kt * 32, mlo, khg);
    v16bf a1 = loadA_lds(lds1, WIDTH, 1, kt * 32, mlo, khg);
#pragma unroll
    for (int i = 0; i < 4; ++i) {
      v16bf bf = loadB_frag(W1f, kt * 32 + (w * 4 + i), lane);
      acc[0][i] = WMMA_BF16(a0, bf, acc[0][i]);
      acc[1][i] = WMMA_BF16(a1, bf, acc[1][i]);
    }
  }
#pragma unroll
  for (int i = 0; i < 4; ++i) {
    int col = (w * 4 + i) * 16 + mlo;
    float bias = b1[col];
#pragma unroll
    for (int v = 0; v < 8; ++v) {
      float x = acc[0][i][v] + bias; x = x > 0.f ? x : 0.f;
      lds0[(v + 8 * khg) * WIDTH + col] = f2bf(x);
      float y = acc[1][i][v] + bias; y = y > 0.f ? y : 0.f;
      lds0[(16 + v + 8 * khg) * WIDTH + col] = f2bf(y);
    }
  }
  __syncthreads();

  // ---------------- layer 2: [32x512] x [512x16] (wave 0 only) ----------------
  if (w == 0) {
    v8f o0 = (v8f)0.f, o1 = (v8f)0.f;
    for (int kt = 0; kt < 16; ++kt) {
      v16bf a0 = loadA_lds(lds0, WIDTH, 0, kt * 32, mlo, khg);
      v16bf a1 = loadA_lds(lds0, WIDTH, 1, kt * 32, mlo, khg);
      v16bf bf = loadB_frag(W2f, kt, lane);   // NT=1
      o0 = WMMA_BF16(a0, bf, o0);
      o1 = WMMA_BF16(a1, bf, o1);
    }
    float bias = b2[mlo];
#pragma unroll
    for (int v = 0; v < 8; ++v) {
      out[(size_t)(rowBase + v + 8 * khg) * OUTC + mlo]      = o0[v] + bias;
      out[(size_t)(rowBase + 16 + v + 8 * khg) * OUTC + mlo] = o1[v] + bias;
    }
  }
}

// ---------------- launch ----------------
extern "C" void kernel_launch(void* const* d_in, const int* in_sizes, int n_in,
                              void* d_out, int out_size, void* d_ws, size_t ws_size,
                              hipStream_t stream) {
  const float* x_target = (const float*)d_in[0];
  const float* x_frozen = (const float*)d_in[1];
  const float* cond     = (const float*)d_in[2];
  const float* W0 = (const float*)d_in[3];
  const float* b0 = (const float*)d_in[4];
  const float* W1 = (const float*)d_in[5];
  const float* b1 = (const float*)d_in[6];
  const float* W2 = (const float*)d_in[7];
  const float* b2 = (const float*)d_in[8];
  float* out = (float*)d_out;

  // workspace layout (bytes): F | W0f | W1f | W2f
  char* ws = (char*)d_ws;
  __bf16* F   = (__bf16*)ws;                                  // 8 MB
  __bf16* W0f = (__bf16*)(ws + (size_t)NROWS * FEAT_P * 2);   // 64 KB
  __bf16* W1f = (__bf16*)((char*)W0f + (size_t)NW0 * 2);      // 512 KB
  __bf16* W2f = (__bf16*)((char*)W1f + (size_t)NW1 * 2);      // 16 KB

  prep_feats<<<(NROWS * FEAT_P) / 256, 256, 0, stream>>>(x_target, x_frozen, cond, F);
  prep_weights<<<(NW0 + NW1 + NW2 + 255) / 256, 256, 0, stream>>>(W0, W1, W2, W0f, W1f, W2f);
  made_mlp<<<NROWS / MBLK, 256, 0, stream>>>(F, W0f, W1f, W2f, b0, b1, b2, out);
}
